// LModule_13786845020430
// MI455X (gfx1250) — matrix operand order repaired and checked by reading, not compiled, not run on hardware
//
#include <hip/hip_runtime.h>
#include <math.h>

// ---------------- problem constants (from reference) ----------------
constexpr int Bc  = 2;
constexpr int Lc  = 2048;
constexpr int Dc  = 1024;
constexpr int Hc  = 16;
constexpr int HDc = 64;          // head dim
constexpr int Mr  = Bc * Lc;     // 4096 rows
constexpr float EPSc = 1e-6f;
constexpr float ROPE_BASEc = 10000.0f;

// ---------------- WMMA types (CDNA5 gfx1250, wave32) ----------------
typedef __attribute__((ext_vector_type(16))) __bf16 v16bf;
typedef __attribute__((ext_vector_type(8)))  float  v8f;

union Frag16 {
    v16bf v;
    unsigned int   u32[8];
    unsigned short u16[16];
};

__device__ __forceinline__ unsigned short f2bf(float f) {
    unsigned int u = __float_as_uint(f);
    unsigned int r = u + 0x7FFFu + ((u >> 16) & 1u);  // round-to-nearest-even
    return (unsigned short)(r >> 16);
}

// A-fragment K mapping (16-bit A 16x32 layout, ISA 7.12.2):
// VGPR j holds K pair {kmap(j), kmap(j)+1} (+8 for upper half-wave)
__device__ __forceinline__ int kmap(int j) { return (j < 4) ? 2 * j : 16 + 2 * (j - 4); }

// ---- gfx1250 async global->LDS path (ASYNCcnt), guarded with fallback ----
#if defined(__HIP_DEVICE_COMPILE__) && \
    __has_builtin(__builtin_amdgcn_global_load_async_to_lds_b128) && \
    __has_builtin(__builtin_amdgcn_s_wait_asynccnt)
#define USE_ASYNC_LDS 1
typedef __attribute__((__vector_size__(4 * sizeof(int)))) int v4i;
typedef __attribute__((address_space(1))) v4i gv4i;   // global int4
typedef __attribute__((address_space(3))) v4i lv4i;   // LDS int4
#else
#define USE_ASYNC_LDS 0
#endif

// ---------------- f32 -> bf16 bulk conversion (weights, done once) --------
__global__ __launch_bounds__(256) void f32_to_bf16_kernel(
    const float* __restrict__ src, unsigned short* __restrict__ dst, int n4)
{
    const int i = blockIdx.x * blockDim.x + threadIdx.x;
    if (i >= n4) return;
    float4 f = ((const float4*)src)[i];
    unsigned short t[4] = { f2bf(f.x), f2bf(f.y), f2bf(f.z), f2bf(f.w) };
    ((uint2*)dst)[i] = *(const uint2*)t;
}

// ---------------- fused add + RMSNorm ----------------
template <bool ADD>
__global__ __launch_bounds__(256) void rmsnorm_kernel(
    const float* __restrict__ a, const float* __restrict__ b,
    const float* __restrict__ g, float* __restrict__ xout,
    unsigned short* __restrict__ hout)
{
    const int row = blockIdx.x;
    const int tid = threadIdx.x;
    const size_t base = (size_t)row * Dc;

    float4 xa = ((const float4*)(a + base))[tid];
    if (ADD) {
        float4 xb = ((const float4*)(b + base))[tid];
        xa.x += xb.x; xa.y += xb.y; xa.z += xb.z; xa.w += xb.w;
    }
    float ss = xa.x * xa.x + xa.y * xa.y + xa.z * xa.z + xa.w * xa.w;
#pragma unroll
    for (int m = 16; m >= 1; m >>= 1) ss += __shfl_xor(ss, m, 32);

    __shared__ float red[8];
    if ((tid & 31) == 0) red[tid >> 5] = ss;
    __syncthreads();
    float tot = 0.f;
#pragma unroll
    for (int i = 0; i < 8; ++i) tot += red[i];

    const float inv = rsqrtf(tot / (float)Dc + EPSc);
    float4 gv = ((const float4*)g)[tid];
    if (ADD) ((float4*)(xout + base))[tid] = xa;
    const int c = tid * 4;
    hout[base + c + 0] = f2bf(xa.x * inv * gv.x);
    hout[base + c + 1] = f2bf(xa.y * inv * gv.y);
    hout[base + c + 2] = f2bf(xa.z * inv * gv.z);
    hout[base + c + 3] = f2bf(xa.w * inv * gv.w);
}

// ---------------- WMMA GEMM: C[M,N] = A[M,K](bf16) @ W[N,K](bf16)^T ------
// Double-buffered LDS, async global->LDS staging, 8 WMMAs per barrier.
// MODE 0: store f32 ; MODE 1: store f32 + resid ; MODE 2: silu -> bf16
constexpr int BM = 128, BN = 64, BK = 64, LDT = 72;  // LDS tile stride (pad)

template <int MODE>
__global__ __launch_bounds__(256) void gemm_bf16_wmma(
    const unsigned short* __restrict__ A,   // [M,K] bf16 row-major
    const unsigned short* __restrict__ W,   // [N,K] bf16 row-major
    const float* __restrict__ resid,        // [M,N] f32 or nullptr
    void* __restrict__ outp,
    int M, int N, int K)
{
    __shared__ unsigned short Abuf[2][BM * LDT];
    __shared__ unsigned short Wbuf[2][BN * LDT];

    const int tid  = threadIdx.x;
    const int lane = tid & 31;
    const int wave = tid >> 5;
    const int hi   = lane >> 4;
    const int ln   = lane & 15;
    const int wm   = wave & 3;    // 4 waves along M (32 rows each)
    const int wn   = wave >> 2;   // 2 waves along N (32 cols each)
    const int rowBase = blockIdx.y * BM;
    const int colBase = blockIdx.x * BN;

    v8f acc[2][2];
#pragma unroll
    for (int i = 0; i < 2; ++i)
#pragma unroll
        for (int j = 0; j < 2; ++j)
#pragma unroll
            for (int r = 0; r < 8; ++r) acc[i][j][r] = 0.f;

    // stage one BKx{BM,BN} tile pair: 6 x b128 per thread-position
    // (A: 128x64 ushort = 4 chunks/thread, W: 64x64 = 2 chunks/thread)
    auto stage = [&](int k0, int buf) {
#pragma unroll
        for (int i = 0; i < 4; ++i) {
            const int id = tid + 256 * i;
            const int r = id >> 3, c = (id & 7) * 8;
            const unsigned short* g = A + (size_t)(rowBase + r) * K + k0 + c;
            unsigned short* l = &Abuf[buf][r * LDT + c];
#if USE_ASYNC_LDS
            __builtin_amdgcn_global_load_async_to_lds_b128((gv4i*)g, (lv4i*)l, 0, 0);
#else
            *(uint4*)l = *(const uint4*)g;
#endif
        }
#pragma unroll
        for (int i = 0; i < 2; ++i) {
            const int id = tid + 256 * i;
            const int r = id >> 3, c = (id & 7) * 8;
            const unsigned short* g = W + (size_t)(colBase + r) * K + k0 + c;
            unsigned short* l = &Wbuf[buf][r * LDT + c];
#if USE_ASYNC_LDS
            __builtin_amdgcn_global_load_async_to_lds_b128((gv4i*)g, (lv4i*)l, 0, 0);
#else
            *(uint4*)l = *(const uint4*)g;
#endif
        }
    };

    stage(0, 0);                        // prologue: tile 0 -> buf 0
    const int ntile = K / BK;
    for (int t = 0; t < ntile; ++t) {
        const int buf = t & 1;
        if (t + 1 < ntile) {
            stage((t + 1) * BK, buf ^ 1);   // prefetch next tile
#if USE_ASYNC_LDS
            __builtin_amdgcn_s_wait_asynccnt(6);  // current tile's 6 done (in-order)
#endif
        } else {
#if USE_ASYNC_LDS
            __builtin_amdgcn_s_wait_asynccnt(0);
#endif
        }
        __syncthreads();

#pragma unroll
        for (int dc = 0; dc < 2; ++dc) {        // two K=32 sub-chunks
            Frag16 af[2], bfm[2];
#pragma unroll
            for (int i = 0; i < 2; ++i) {       // A fragments (16x32, A layout)
                const int row = wm * 32 + i * 16 + ln;
#pragma unroll
                for (int j = 0; j < 8; ++j) {
                    const int k = dc * 32 + kmap(j) + hi * 8;
                    af[i].u32[j] = *(const unsigned int*)&Abuf[buf][row * LDT + k];
                }
            }
#pragma unroll
            for (int i = 0; i < 2; ++i) {       // B fragments (32x16, B layout)
                const int row = wn * 32 + i * 16 + ln;
#pragma unroll
                for (int j = 0; j < 8; ++j) {
                    const int k = dc * 32 + kmap(j) + hi * 8;
                    bfm[i].u32[j] = *(const unsigned int*)&Wbuf[buf][row * LDT + k];
                }
            }
#pragma unroll
            for (int i = 0; i < 2; ++i)
#pragma unroll
                for (int j = 0; j < 2; ++j)
                    acc[i][j] = __builtin_amdgcn_wmma_f32_16x16x32_bf16(
                        false, af[i].v, false, bfm[j].v, (short)0, acc[i][j], false, false);
        }
        __syncthreads();
    }

    // epilogue: C layout -> row = r + 8*hi, col = ln (per 16x16 tile)
#pragma unroll
    for (int i = 0; i < 2; ++i)
#pragma unroll
        for (int j = 0; j < 2; ++j)
#pragma unroll
            for (int r = 0; r < 8; ++r) {
                const int row = rowBase + wm * 32 + i * 16 + r + hi * 8;
                const int col = colBase + wn * 32 + j * 16 + ln;
                const size_t idx = (size_t)row * N + col;
                const float v = acc[i][j][r];
                if constexpr (MODE == 0) {
                    ((float*)outp)[idx] = v;
                } else if constexpr (MODE == 1) {
                    ((float*)outp)[idx] = v + resid[idx];
                } else {
                    const float sv = v / (1.0f + __expf(-v));   // SiLU
                    ((unsigned short*)outp)[idx] = f2bf(sv);
                }
            }
}

// ---------------- RoPE + qkv split to bf16 (B,H,L,HD) ----------------
__global__ __launch_bounds__(256) void rope_split_kernel(
    const float* __restrict__ qkv,
    unsigned short* __restrict__ qb,
    unsigned short* __restrict__ kb,
    unsigned short* __restrict__ vb)
{
    const int idx = blockIdx.x * blockDim.x + threadIdx.x;  // B*H*L*32
    const int d  = idx & 31;            // rotation pair index (half = 32)
    const int t  = idx >> 5;            // (b*H + h)*L + l
    const int l  = t & (Lc - 1);
    const int bh = t >> 11;             // log2(L) = 11
    const int h  = bh & (Hc - 1);
    const int b  = bh >> 4;

    const size_t base = ((size_t)(b * Lc + l)) * (3 * Dc) + h * HDc;
    const float q1 = qkv[base + d],          q2 = qkv[base + d + 32];
    const float k1 = qkv[base + Dc + d],     k2 = qkv[base + Dc + d + 32];
    const float v1 = qkv[base + 2 * Dc + d], v2 = qkv[base + 2 * Dc + d + 32];

    const float invf = __powf(ROPE_BASEc, -(float)d * (1.0f / 32.0f));
    const float fr = (float)l * invf;
    float sn, cs;
    __sincosf(fr, &sn, &cs);

    const size_t ob = ((size_t)bh * Lc + l) * HDc;
    qb[ob + d]      = f2bf(q1 * cs - q2 * sn);
    qb[ob + d + 32] = f2bf(q2 * cs + q1 * sn);
    kb[ob + d]      = f2bf(k1 * cs - k2 * sn);
    kb[ob + d + 32] = f2bf(k2 * cs + k1 * sn);
    vb[ob + d]      = f2bf(v1);
    vb[ob + d + 32] = f2bf(v2);
}

// ---------------- flash attention (online softmax, WMMA bf16) ----------------
// 128 threads = 4 waves; each wave owns 16 query rows; 32 keys per iteration.
__global__ __launch_bounds__(128) void attn_kernel(
    const unsigned short* __restrict__ qb,
    const unsigned short* __restrict__ kb,
    const unsigned short* __restrict__ vb,
    unsigned short* __restrict__ out)
{
    __shared__ unsigned short Pbuf[4][16][36];   // per-wave P tile (16q x 32k)

    const int tid  = threadIdx.x;
    const int lane = tid & 31;
    const int wave = tid >> 5;
    const int hi   = lane >> 4;
    const int ln   = lane & 15;

    const int nqc = Lc / 64;
    const int bh  = blockIdx.x / nqc;
    const int qc  = blockIdx.x % nqc;
    const int qbase = qc * 64 + wave * 16;
    const int b = bh >> 4, h = bh & 15;

    // Q fragments: two 16x32 A-fragments covering HD=64
    Frag16 qf[2];
#pragma unroll
    for (int dc = 0; dc < 2; ++dc)
#pragma unroll
        for (int j = 0; j < 8; ++j) {
            const int k = kmap(j) + hi * 8;
            qf[dc].u32[j] = *(const unsigned int*)
                &qb[((size_t)bh * Lc + qbase + ln) * HDc + dc * 32 + k];
        }

    v8f o[4];
    float mrow[8], lrow[8];
#pragma unroll
    for (int t4 = 0; t4 < 4; ++t4)
#pragma unroll
        for (int r = 0; r < 8; ++r) o[t4][r] = 0.f;
#pragma unroll
    for (int r = 0; r < 8; ++r) { mrow[r] = -1e30f; lrow[r] = 0.f; }

    for (int k0 = 0; k0 < Lc; k0 += 32) {
        v8f s0, s1;
#pragma unroll
        for (int r = 0; r < 8; ++r) { s0[r] = 0.f; s1[r] = 0.f; }

        // S = Q K^T : contraction over HD in two 32-chunks
#pragma unroll
        for (int dc = 0; dc < 2; ++dc) {
            Frag16 kf0, kf1;
#pragma unroll
            for (int j = 0; j < 8; ++j) {
                const int kk = kmap(j) + hi * 8;        // contraction (d) index
                kf0.u32[j] = *(const unsigned int*)
                    &kb[((size_t)bh * Lc + k0 + ln) * HDc + dc * 32 + kk];
                kf1.u32[j] = *(const unsigned int*)
                    &kb[((size_t)bh * Lc + k0 + 16 + ln) * HDc + dc * 32 + kk];
            }
            s0 = __builtin_amdgcn_wmma_f32_16x16x32_bf16(false, qf[dc].v, false, kf0.v, (short)0, s0, false, false);
            s1 = __builtin_amdgcn_wmma_f32_16x16x32_bf16(false, qf[dc].v, false, kf1.v, (short)0, s1, false, false);
        }

        // online softmax: row stats distributed per C layout (row = r + 8*hi)
        float alphas[8];
#pragma unroll
        for (int r = 0; r < 8; ++r) {
            const float sv0 = s0[r] * 0.125f;   // 1/sqrt(HD)
            const float sv1 = s1[r] * 0.125f;
            float t = fmaxf(sv0, sv1);
#pragma unroll
            for (int m = 8; m >= 1; m >>= 1) t = fmaxf(t, __shfl_xor(t, m, 16));
            const float mn = fmaxf(mrow[r], t);
            const float p0 = __expf(sv0 - mn);
            const float p1 = __expf(sv1 - mn);
            float rs = p0 + p1;
#pragma unroll
            for (int m = 8; m >= 1; m >>= 1) rs += __shfl_xor(rs, m, 16);
            const float al = __expf(mrow[r] - mn);
            lrow[r] = lrow[r] * al + rs;
            mrow[r] = mn;
            alphas[r] = al;
            Pbuf[wave][r + hi * 8][ln]      = f2bf(p0);
            Pbuf[wave][r + hi * 8][16 + ln] = f2bf(p1);
        }
#pragma unroll
        for (int t4 = 0; t4 < 4; ++t4)
#pragma unroll
            for (int r = 0; r < 8; ++r) o[t4][r] *= alphas[r];
        __syncthreads();

        // P as A-fragment (16x32) from LDS
        Frag16 pf;
#pragma unroll
        for (int j = 0; j < 8; ++j) {
            const int k = kmap(j) + hi * 8;
            pf.u32[j] = *(const unsigned int*)&Pbuf[wave][ln][k];
        }

        // O += P V : 4 N-tiles of 16 over HD
#pragma unroll
        for (int t4 = 0; t4 < 4; ++t4) {
            Frag16 vf;
#pragma unroll
            for (int j = 0; j < 8; ++j) {
                const int kk = kmap(j) + hi * 8;        // key index (strided in mem)
                vf.u16[2 * j]     = vb[((size_t)bh * Lc + k0 + kk)     * HDc + t4 * 16 + ln];
                vf.u16[2 * j + 1] = vb[((size_t)bh * Lc + k0 + kk + 1) * HDc + t4 * 16 + ln];
            }
            o[t4] = __builtin_amdgcn_wmma_f32_16x16x32_bf16(false, pf.v, false, vf.v, (short)0, o[t4], false, false);
        }
        __syncthreads();
    }

    // finalize: divide by l, store bf16 at (b, l, h*HD + d) for out-proj GEMM
#pragma unroll
    for (int t4 = 0; t4 < 4; ++t4)
#pragma unroll
        for (int r = 0; r < 8; ++r) {
            const int q = qbase + r + hi * 8;
            const float val = o[t4][r] / lrow[r];
            out[((size_t)(b * Lc + q)) * Dc + h * HDc + t4 * 16 + ln] = f2bf(val);
        }
}

// ---------------- launcher ----------------
extern "C" void kernel_launch(void* const* d_in, const int* in_sizes, int n_in,
                              void* d_out, int out_size, void* d_ws, size_t ws_size,
                              hipStream_t stream)
{
    (void)in_sizes; (void)n_in; (void)out_size; (void)ws_size;
    const float* zH     = (const float*)d_in[0];
    const float* zL     = (const float*)d_in[1];
    const float* w_qkv  = (const float*)d_in[2];
    const float* w_out  = (const float*)d_in[3];
    const float* w_ffn1 = (const float*)d_in[4];
    const float* w_ffn2 = (const float*)d_in[5];
    const float* g1     = (const float*)d_in[6];
    const float* g2     = (const float*)d_in[7];

    char* ws = (char*)d_ws;
    const size_t MB = 1ull << 20;
    float*          X   = (float*)(ws + 0);                // 16MB  x = zH+zL
    unsigned short* Hb  = (unsigned short*)(ws + 16*MB);   // 8MB   h / h2 (bf16)
    float*          QKV = (float*)(ws + 24*MB);            // 48MB  qkv f32
    float*          X2  = (float*)(ws + 24*MB);            // 16MB  (reuses dead QKV)
    unsigned short* U   = (unsigned short*)(ws + 40*MB);   // 32MB  ffn hidden (reuses QKV)
    unsigned short* QB  = (unsigned short*)(ws + 72*MB);   // 8MB   roped Q bf16
    unsigned short* KB  = (unsigned short*)(ws + 80*MB);   // 8MB   roped K bf16
    unsigned short* VB  = (unsigned short*)(ws + 88*MB);   // 8MB   V bf16
    unsigned short* AO  = (unsigned short*)(ws + 96*MB);   // 8MB   attention out bf16
    unsigned short* WQ  = (unsigned short*)(ws + 104*MB);  // 6MB   w_qkv  bf16
    unsigned short* WO  = (unsigned short*)(ws + 110*MB);  // 2MB   w_out  bf16
    unsigned short* W1  = (unsigned short*)(ws + 112*MB);  // 8MB   w_ffn1 bf16
    unsigned short* W2  = (unsigned short*)(ws + 120*MB);  // 8MB   w_ffn2 bf16

    // 0) pre-convert all weights to bf16 (halves per-GEMM weight traffic)
    {
        const int nq = 3 * Dc * Dc / 4, no = Dc * Dc / 4, nf = 4 * Dc * Dc / 4;
        f32_to_bf16_kernel<<<nq / 256, 256, 0, stream>>>(w_qkv,  WQ, nq);
        f32_to_bf16_kernel<<<no / 256, 256, 0, stream>>>(w_out,  WO, no);
        f32_to_bf16_kernel<<<nf / 256, 256, 0, stream>>>(w_ffn1, W1, nf);
        f32_to_bf16_kernel<<<nf / 256, 256, 0, stream>>>(w_ffn2, W2, nf);
    }
    // 1) x = zH + zL ; h = rmsnorm(x, g1)
    rmsnorm_kernel<true><<<Mr, 256, 0, stream>>>(zH, zL, g1, X, Hb);
    // 2) qkv = h @ w_qkv^T
    gemm_bf16_wmma<0><<<dim3(3 * Dc / BN, Mr / BM), 256, 0, stream>>>(
        Hb, WQ, nullptr, QKV, Mr, 3 * Dc, Dc);
    // 3) RoPE(q,k) + split to (B,H,L,HD) bf16
    rope_split_kernel<<<(Bc * Hc * Lc * 32) / 256, 256, 0, stream>>>(QKV, QB, KB, VB);
    // 4) attention (flash, online softmax)
    attn_kernel<<<Bc * Hc * (Lc / 64), 128, 0, stream>>>(QB, KB, VB, AO);
    // 5) x2 = x + attn @ w_out^T
    gemm_bf16_wmma<1><<<dim3(Dc / BN, Mr / BM), 256, 0, stream>>>(
        AO, WO, X, X2, Mr, Dc, Dc);
    // 6) h2 = rmsnorm(x2, g2)
    rmsnorm_kernel<false><<<Mr, 256, 0, stream>>>(X2, nullptr, g2, nullptr, Hb);
    // 7) u = silu(h2 @ w_ffn1^T)  -> bf16
    gemm_bf16_wmma<2><<<dim3(4 * Dc / BN, Mr / BM), 256, 0, stream>>>(
        Hb, W1, nullptr, U, Mr, 4 * Dc, Dc);
    // 8) out = x2 + u @ w_ffn2^T
    gemm_bf16_wmma<1><<<dim3(Dc / BN, Mr / BM), 256, 0, stream>>>(
        U, W2, X2, (float*)d_out, Mr, Dc, 4 * Dc);
}
